// SwinUnet_10256381903563
// MI455X (gfx1250) — matrix operand order repaired and checked by reading, not compiled, not run on hardware
//
#include <hip/hip_runtime.h>
#include <hip/hip_bf16.h>

// MI455X / gfx1250 fused Swin window-attention.
//   kernel 1: transpose+convert w_qkv, w_out -> bf16 in d_ws (B-operand friendly)
//   kernel 2: one workgroup per window (2048 blocks, 256 thr = 8 waves):
//     gather shifted window -> LDS bf16 (bank-conflict-free padded rows)
//     GEMM1 (x@Wqkv) via v_wmma_f32_16x16x32_bf16 -> Q,K,Vt in LDS
//     per-head attention: S=Q@K^T (+bias+mask), shfl-xor softmax, P@V
//     GEMM2 (o@Wout) -> scattered fp32 store (window reverse + roll)

#define IMG   64
#define HEADS 12
#define CH    384
#define D     64
#define E     32
#define K3C   1152
#define NWIN  2048

// Padded LDS row strides (in bf16 elements). Chosen so the dword residue
// mod 64 banks spreads 16 lanes over all banks:
//   XSTR: 784B = 196 dw == 4 (mod 64)  -> 4*l, conflict-free for b128 loads
//   QKSTR: 80B =  20 dw == 4*(5l%16)   -> uniform 2 accesses/bank for 32B/lane
//   VSTR: 144B =  36 dw == 4*(9l%16)   -> uniform 2 accesses/bank for 32B/lane
#define XSTR  392
#define QKSTR 40
#define VSTR  72
#define PSTR  72

typedef __attribute__((ext_vector_type(16))) __bf16 v16bf;
typedef __attribute__((ext_vector_type(8)))  __bf16 v8bf;
typedef __attribute__((ext_vector_type(8)))  float  v8f;
typedef __attribute__((ext_vector_type(4)))  float  v4f;

static __device__ __forceinline__ __bf16 tobf(float f) {
  unsigned u = __builtin_bit_cast(unsigned, f);
  u += 0x7FFFu + ((u >> 16) & 1u);                   // round-to-nearest-even
  unsigned short h = (unsigned short)(u >> 16);
  return __builtin_bit_cast(__bf16, h);
}
static __device__ __forceinline__ unsigned pack2(float a, float b) {
  unsigned short lo = __builtin_bit_cast(unsigned short, tobf(a));
  unsigned short hi = __builtin_bit_cast(unsigned short, tobf(b));
  return (unsigned)lo | ((unsigned)hi << 16);
}
static __device__ __forceinline__ int lab(int i) { return (i < 56) ? 0 : ((i < 60) ? 1 : 2); }

// A-fragment (16x32 bf16): lane<16 -> row m0+lane, k = kk+{0..15}
//                          lane>=16 -> row m0+lane-16, k = kk+{16..31}
static __device__ __forceinline__ v16bf load_a16x32(const __bf16* base, int stride,
                                                    int m0, int kk, int lane) {
  const __bf16* p = base + (m0 + (lane & 15)) * stride + kk + ((lane >> 4) << 4);
  v8bf lo = *(const v8bf*)p;
  v8bf hi = *(const v8bf*)(p + 8);
  return __builtin_shufflevector(lo, hi, 0,1,2,3,4,5,6,7,8,9,10,11,12,13,14,15);
}
// B-fragment: per-lane 16 contiguous bf16 (lanes 0-15: K=0..15, 16-31: K=16..31)
// 32B-aligned source (global weight rows): single vector load
static __device__ __forceinline__ v16bf load_b_row(const __bf16* row, int lane) {
  return *(const v16bf*)(row + ((lane >> 4) << 4));
}
// 16B-aligned source (padded LDS rows): two 16B halves
static __device__ __forceinline__ v16bf load_b16(const __bf16* p) {
  v8bf lo = *(const v8bf*)p;
  v8bf hi = *(const v8bf*)(p + 8);
  return __builtin_shufflevector(lo, hi, 0,1,2,3,4,5,6,7,8,9,10,11,12,13,14,15);
}
static __device__ __forceinline__ v8f wmma_bf16(v16bf a, v16bf b, v8f c) {
  return __builtin_amdgcn_wmma_f32_16x16x32_bf16(false, a, false, b, (short)0, c, false, false);
}

__global__ void __launch_bounds__(256)
prep_weights(const float* __restrict__ w_qkv, const float* __restrict__ w_out,
             __bf16* __restrict__ tq, __bf16* __restrict__ to) {
  int i = blockIdx.x * 256 + threadIdx.x;
  if (i < K3C * CH) {                       // tq[m][c] = w_qkv[c][m]
    int m = i / CH, c = i - m * CH;
    tq[i] = tobf(w_qkv[(size_t)c * K3C + m]);
  } else {
    int j = i - K3C * CH;
    if (j < CH * CH) {                      // to[m][c] = w_out[c][m]
      int m = j / CH, c = j - m * CH;
      to[j] = tobf(w_out[(size_t)c * CH + m]);
    }
  }
}

__global__ void __launch_bounds__(256, 1)
swin_fused(const float* __restrict__ x,
           const float* __restrict__ b_qkv,
           const float* __restrict__ b_out,
           const float* __restrict__ bias_table,
           const __bf16* __restrict__ wT_qkv,   // [1152][384] bf16
           const __bf16* __restrict__ wT_out,   // [384][384]  bf16
           float* __restrict__ out) {
  __shared__ __bf16 sXW[D * XSTR];          // window tokens; reused for attn output
  __shared__ __bf16 sQ [HEADS * D * QKSTR]; // [h][tok][e]  (pre-scaled)
  __shared__ __bf16 sK [HEADS * D * QKSTR]; // [h][tok][e]
  __shared__ __bf16 sVt[HEADS * E * VSTR];  // [h][e][tok]  (transposed)
  __shared__ __bf16 sP [8 * 16 * PSTR];     // per-wave P strip

  const int n    = blockIdx.x;
  const int bat  = n >> 6;
  const int wwi  = (n >> 3) & 7;
  const int wwj  = n & 7;
  const int tid  = threadIdx.x;
  // wave id is uniform across the wave: pin it to an SGPR so all derived loop
  // bounds / tile selectors scalarize (no divergent EXEC-mask branches).
  const int wave = __builtin_amdgcn_readfirstlane(tid >> 5);
  const int lane = tid & 31;
  const int l15  = lane & 15;
  const int lh   = (lane >> 4) << 3;        // +8 row offset for upper half-wave
  const int bsel = (lane >> 4) << 4;        // +16 K offset for B fragments

  // ---- Phase 0: gather shifted window (roll -4,-4) into LDS as bf16
  {
    int t  = tid >> 2;
    int cg = (tid & 3) * 96;
    int gi = (wwi * 8 + (t >> 3) + 4) & 63;
    int gj = (wwj * 8 + (t & 7)  + 4) & 63;
    const float* src = x + ((size_t)(bat * IMG + gi) * IMG + gj) * CH + cg;
    unsigned* dst = (unsigned*)&sXW[t * XSTR + cg];
#pragma unroll
    for (int i = 0; i < 96; i += 4) {
      v4f v = *(const v4f*)(src + i);
      dst[(i >> 1)    ] = pack2(v.x, v.y);
      dst[(i >> 1) + 1] = pack2(v.z, v.w);
    }
  }
  __syncthreads();

  // ---- Phase 1: QKV GEMM  [64x384]@[384x1152], 9 N-tiles per wave
  // k-loop fully unrolled: scheduler issues B loads far ahead, waits
  // incrementally; no register-rotation copies, no hazard NOPs.
  const float scale = 0.17677669529663687f;   // 32^-0.5
#pragma unroll 1
  for (int nj = wave * 9; nj < wave * 9 + 9; ++nj) {
    int ncol = nj * 16 + l15;
    const __bf16* brow = wT_qkv + (size_t)ncol * CH;
    float bq = b_qkv[ncol];
    v8f acc[4] = {};
#pragma unroll
    for (int kk = 0; kk < CH; kk += 32) {
      v16bf b = load_b_row(brow + kk, lane);
#pragma unroll
      for (int mi = 0; mi < 4; ++mi)
        acc[mi] = wmma_bf16(load_a16x32(sXW, XSTR, mi * 16, kk, lane), b, acc[mi]);
    }
    // Branchless epilogue: nj is scalar, so s/h select a base+stride once.
    int s  = nj / 24;                 // 0=Q 1=K 2=V
    int rm = nj % 24;
    int h  = rm >> 1;
    int e  = ((rm & 1) << 4) + l15;
    __bf16* dst;
    int  tokstride;
    float mulf;
    if (s == 0)      { dst = sQ  + h * D * QKSTR + e;   tokstride = QKSTR; mulf = scale; }
    else if (s == 1) { dst = sK  + h * D * QKSTR + e;   tokstride = QKSTR; mulf = 1.0f;  }
    else             { dst = sVt + (h * E + e) * VSTR;  tokstride = 1;     mulf = 1.0f;  }
#pragma unroll
    for (int mi = 0; mi < 4; ++mi) {
#pragma unroll
      for (int r = 0; r < 8; ++r) {
        int tok = mi * 16 + r + lh;
        dst[tok * tokstride] = tobf((acc[mi][r] + bq) * mulf);
      }
    }
  }
  __syncthreads();

  // ---- Phase 2: attention, 48 (head, 16-row strip) tasks, 6 per wave
#pragma unroll 1
  for (int task = wave * 6; task < wave * 6 + 6; ++task) {
    int h  = task >> 2;               // scalar
    int mi = task & 3;                // scalar
    const __bf16* Qh = sQ  + h * D * QKSTR;
    const __bf16* Kh = sK  + h * D * QKSTR;
    const __bf16* Vh = sVt + h * E * VSTR;
    v16bf aq = load_a16x32(Qh, QKSTR, mi * 16, 0, lane);
    v8f s4[4];
#pragma unroll
    for (int nj = 0; nj < 4; ++nj) {
      v16bf bk = load_b16(Kh + (nj * 16 + l15) * QKSTR + bsel);
      v8f z = {};
      s4[nj] = wmma_bf16(aq, bk, z);
    }
    // relative-position bias (idx = di+dj+29) + shift mask (region labels)
#pragma unroll
    for (int nj = 0; nj < 4; ++nj) {
      int ktok = nj * 16 + l15;
      int ki = ktok >> 3, kj = ktok & 7;
      int klab = lab(wwi * 8 + ki) * 3 + lab(wwj * 8 + kj);
#pragma unroll
      for (int r = 0; r < 8; ++r) {
        int qtok = mi * 16 + r + lh;
        int qi = qtok >> 3, qj = qtok & 7;
        float bv = bias_table[((qi - ki) + (qj - kj) + 29) * HEADS + h];
        int qlab = lab(wwi * 8 + qi) * 3 + lab(wwj * 8 + qj);
        s4[nj][r] += bv + ((qlab == klab) ? 0.f : -100.f);
      }
    }
    // softmax over 64 cols: elementwise over 4 tiles + 16-lane shfl_xor reduce
    __bf16* Pw = sP + wave * 16 * PSTR;
#pragma unroll
    for (int r = 0; r < 8; ++r) {
      float m = fmaxf(fmaxf(s4[0][r], s4[1][r]), fmaxf(s4[2][r], s4[3][r]));
      m = fmaxf(m, __shfl_xor(m, 1, 32));
      m = fmaxf(m, __shfl_xor(m, 2, 32));
      m = fmaxf(m, __shfl_xor(m, 4, 32));
      m = fmaxf(m, __shfl_xor(m, 8, 32));
      float e0 = __expf(s4[0][r] - m);
      float e1 = __expf(s4[1][r] - m);
      float e2 = __expf(s4[2][r] - m);
      float e3 = __expf(s4[3][r] - m);
      float sum = e0 + e1 + e2 + e3;
      sum += __shfl_xor(sum, 1, 32);
      sum += __shfl_xor(sum, 2, 32);
      sum += __shfl_xor(sum, 4, 32);
      sum += __shfl_xor(sum, 8, 32);
      float inv = 1.0f / sum;
      int row = r + lh;
      Pw[row * PSTR +  0 + l15] = tobf(e0 * inv);
      Pw[row * PSTR + 16 + l15] = tobf(e1 * inv);
      Pw[row * PSTR + 32 + l15] = tobf(e2 * inv);
      Pw[row * PSTR + 48 + l15] = tobf(e3 * inv);
    }
    // O = P @ V  (same-wave LDS: hardware keeps DS ops in order)
#pragma unroll
    for (int et = 0; et < 2; ++et) {
      v8f o = {};
#pragma unroll
      for (int kb = 0; kb < 64; kb += 32) {
        v16bf a = load_a16x32(Pw, PSTR, 0, kb, lane);
        v16bf b = load_b16(Vh + (et * 16 + l15) * VSTR + kb + bsel);
        o = wmma_bf16(a, b, o);
      }
#pragma unroll
      for (int r = 0; r < 8; ++r) {
        int tok = mi * 16 + r + lh;
        sXW[tok * XSTR + h * E + et * 16 + l15] = tobf(o[r]);
      }
    }
  }
  __syncthreads();

  // ---- Phase 3: output projection [64x384]@[384x384] + scatter (reverse+roll)
#pragma unroll 1
  for (int nj = wave * 3; nj < wave * 3 + 3; ++nj) {
    int ncol = nj * 16 + l15;
    const __bf16* brow = wT_out + (size_t)ncol * CH;
    float bo = b_out[ncol];
    v8f acc[4] = {};
#pragma unroll
    for (int kk = 0; kk < CH; kk += 32) {
      v16bf b = load_b_row(brow + kk, lane);
#pragma unroll
      for (int mi = 0; mi < 4; ++mi)
        acc[mi] = wmma_bf16(load_a16x32(sXW, XSTR, mi * 16, kk, lane), b, acc[mi]);
    }
#pragma unroll
    for (int mi = 0; mi < 4; ++mi) {
#pragma unroll
      for (int r = 0; r < 8; ++r) {
        int tok = mi * 16 + r + lh;
        int gi = (wwi * 8 + (tok >> 3) + 4) & 63;
        int gj = (wwj * 8 + (tok & 7)  + 4) & 63;
        out[((size_t)(bat * IMG + gi) * IMG + gj) * CH + ncol] = acc[mi][r] + bo;
      }
    }
  }
}

extern "C" void kernel_launch(void* const* d_in, const int* in_sizes, int n_in,
                              void* d_out, int out_size, void* d_ws, size_t ws_size,
                              hipStream_t stream) {
  const float* x     = (const float*)d_in[0];
  const float* w_qkv = (const float*)d_in[1];
  const float* b_qkv = (const float*)d_in[2];
  const float* w_out = (const float*)d_in[3];
  const float* b_out = (const float*)d_in[4];
  const float* btab  = (const float*)d_in[5];
  float* out = (float*)d_out;

  __bf16* tq = (__bf16*)d_ws;                      // 1152*384 bf16
  __bf16* to = tq + (size_t)K3C * CH;              // 384*384  bf16

  int total = K3C * CH + CH * CH;
  prep_weights<<<(total + 255) / 256, 256, 0, stream>>>(w_qkv, w_out, tq, to);
  swin_fused<<<NWIN, 256, 0, stream>>>(x, b_qkv, b_out, btab, tq, to, out);
}
